// DiffusionTransformer_34995393528039
// MI455X (gfx1250) — matrix operand order repaired and checked by reading, not compiled, not run on hardware
//
#include <hip/hip_runtime.h>

typedef __bf16 bf16_t;
typedef __attribute__((ext_vector_type(16))) __bf16 v16bf;
typedef __attribute__((ext_vector_type(8)))  float  v8f;

__device__ __forceinline__ bf16_t f2bf(float f) {
  union { float f; unsigned u; } a; a.f = f;
  unsigned r = a.u + 0x7fffu + ((a.u >> 16) & 1u);
  union { unsigned short s; bf16_t b; } c; c.s = (unsigned short)(r >> 16);
  return c.b;
}
__device__ __forceinline__ float bf2f(bf16_t b) {
  union { unsigned short s; bf16_t b; } c; c.b = b;
  union { unsigned u; float f; } a; a.u = ((unsigned)c.s) << 16;
  return a.f;
}
__device__ __forceinline__ float act_apply(float x, int act) {
  if (act == 1) return 1.f / (1.f + __expf(-x));          // sigmoid
  if (act == 2) return x / (1.f + __expf(-x));            // silu
  return x;
}
// Load a 16-element bf16 WMMA fragment as two contiguous 16B LDS reads.
__device__ __forceinline__ v16bf ld_frag(const bf16_t* p0, const bf16_t* p1) {
  union { int4 q[2]; v16bf v; } u;
  u.q[0] = *reinterpret_cast<const int4*>(p0);
  u.q[1] = *reinterpret_cast<const int4*>(p1);
  return u.v;
}

// ---------------------------------------------------------------------------
// Row LayerNorm (no affine) fp32 -> bf16.  do_ln==0: plain convert.
// ---------------------------------------------------------------------------
__global__ __launch_bounds__(256) void row_ln_kernel(
    const float* __restrict__ x, bf16_t* __restrict__ y, int C, int do_ln) {
  const int row = blockIdx.x;
  const float* xr = x + (size_t)row * C;
  float s = 0.f, ss = 0.f;
  for (int c = threadIdx.x; c < C; c += 256) { float v = xr[c]; s += v; ss += v * v; }
  #pragma unroll
  for (int o = 16; o > 0; o >>= 1) { s += __shfl_xor(s, o, 32); ss += __shfl_xor(ss, o, 32); }
  __shared__ float rs[8], rss[8];
  if ((threadIdx.x & 31) == 0) { rs[threadIdx.x >> 5] = s; rss[threadIdx.x >> 5] = ss; }
  __syncthreads();
  float ts = 0.f, tss = 0.f;
  #pragma unroll
  for (int i = 0; i < 8; ++i) { ts += rs[i]; tss += rss[i]; }
  const float mean = ts / (float)C;
  const float var  = tss / (float)C - mean * mean;
  const float inv  = rsqrtf(var + 1e-5f);
  for (int c = threadIdx.x; c < C; c += 256) {
    float v = xr[c];
    y[(size_t)row * C + c] = f2bf(do_ln ? (v - mean) * inv : v);
  }
}

// ---------------------------------------------------------------------------
// Fold ln_z_w/ln_z_b into wpb for all 4 blocks.
// Emits WzT TRANSPOSED: WzT[n][c] (n = blk*16+h in 0..63, c in 0..127), bz[64].
// ---------------------------------------------------------------------------
__global__ void fuse_z_kernel(const float* __restrict__ lnw, const float* __restrict__ lnb,
                              const float* __restrict__ wpb, bf16_t* __restrict__ WzT,
                              float* __restrict__ bz) {
  int n = threadIdx.x;
  if (n >= 64) return;
  int b = n >> 4, hh = n & 15;
  float acc = 0.f;
  for (int c = 0; c < 128; ++c) {
    float w = wpb[((size_t)b * 128 + c) * 16 + hh];
    WzT[(size_t)n * 128 + c] = f2bf(lnw[b * 128 + c] * w);
    acc += lnb[b * 128 + c] * w;
  }
  bz[n] = acc;
}

// ---------------------------------------------------------------------------
// Pair-bias: one pass over z. LN(128ch) per pair -> WMMA 64x64x128 against
// fused WzT -> +bz +beta -> planar P[64][S][S] bf16 (plane n = blk*16+h).
// ---------------------------------------------------------------------------
__global__ __launch_bounds__(256) void pair_bias_kernel(
    const float* __restrict__ z, const float* __restrict__ beta,
    const bf16_t* __restrict__ WzT, const float* __restrict__ bz,
    bf16_t* __restrict__ P, int S) {
  __shared__ __align__(16) bf16_t As[64][144];    // pairs x channels (A, k-minor)
  __shared__ __align__(16) bf16_t Wt[64][136];    // n x k (B transposed, k-minor)
  __shared__ __align__(16) float  Cs[64][68];
  const int tid = threadIdx.x, lane = tid & 31, wid = tid >> 5;
  const int i = blockIdx.y;
  const int j0 = blockIdx.x * 64;

  { // stage WzT (already transposed in global): pure vector copies
    int row = tid >> 2, c0 = (tid & 3) * 32;
    const int4* src = reinterpret_cast<const int4*>(WzT + (size_t)row * 128 + c0);
    int4* dst = reinterpret_cast<int4*>(&Wt[row][c0]);
    dst[0] = src[0]; dst[1] = src[1]; dst[2] = src[2]; dst[3] = src[3];
  }
  { // per-pair LN over 128 channels (4 lanes per pair)
    int p = tid >> 2, q = tid & 3;
    const float* zp = z + ((size_t)i * S + j0 + p) * 128 + q * 32;
    float vals[32];
    float s = 0.f, ss = 0.f;
    #pragma unroll
    for (int k = 0; k < 8; ++k) {
      float4 v4 = reinterpret_cast<const float4*>(zp)[k];
      vals[4*k+0] = v4.x; vals[4*k+1] = v4.y; vals[4*k+2] = v4.z; vals[4*k+3] = v4.w;
      s  += v4.x + v4.y + v4.z + v4.w;
      ss += v4.x*v4.x + v4.y*v4.y + v4.z*v4.z + v4.w*v4.w;
    }
    s += __shfl_xor(s, 1, 32); ss += __shfl_xor(ss, 1, 32);
    s += __shfl_xor(s, 2, 32); ss += __shfl_xor(ss, 2, 32);
    const float mean = s * (1.f / 128.f);
    const float var  = ss * (1.f / 128.f) - mean * mean;
    const float inv  = rsqrtf(var + 1e-5f);
    #pragma unroll
    for (int k = 0; k < 32; ++k) As[p][q * 32 + k] = f2bf((vals[k] - mean) * inv);
  }
  __syncthreads();

  const int wm = wid & 3, wn = wid >> 2;
  const int m = lane & 15, kh = lane >> 4;
  v8f acc0 = {}, acc1 = {};
  #pragma unroll
  for (int k0 = 0; k0 < 128; k0 += 32) {
    v16bf af = ld_frag(&As[wm * 16 + m][k0 + kh * 16], &As[wm * 16 + m][k0 + kh * 16 + 8]);
    v16bf b0 = ld_frag(&Wt[wn * 32 + m][k0 + kh * 16], &Wt[wn * 32 + m][k0 + kh * 16 + 8]);
    v16bf b1 = ld_frag(&Wt[wn * 32 + 16 + m][k0 + kh * 16], &Wt[wn * 32 + 16 + m][k0 + kh * 16 + 8]);
    acc0 = __builtin_amdgcn_wmma_f32_16x16x32_bf16(false, af, false, b0, (short)0, acc0, false, false);
    acc1 = __builtin_amdgcn_wmma_f32_16x16x32_bf16(false, af, false, b1, (short)0, acc1, false, false);
  }
  const int lh = lane >> 4;
  #pragma unroll
  for (int sub = 0; sub < 2; ++sub) {
    v8f acc = sub ? acc1 : acc0;
    int n = wn * 32 + sub * 16 + m;
    float bv = bz[n];
    #pragma unroll
    for (int r = 0; r < 8; ++r) Cs[wm * 16 + r + 8 * lh][n] = acc[r] + bv;
  }
  __syncthreads();
  { // add beta, write planar bf16 planes
    int n = tid >> 2, jseg = (tid & 3) * 16;
    int hh = n & 15;
    union { int4 q[2]; bf16_t b[16]; } pk;
    #pragma unroll
    for (int jj = 0; jj < 16; ++jj) {
      float val = Cs[jseg + jj][n] + beta[((size_t)i * S + j0 + jseg + jj) * 16 + hh];
      pk.b[jj] = f2bf(val);
    }
    size_t base = ((size_t)n * S + i) * S + j0 + jseg;
    reinterpret_cast<int4*>(P + base)[0] = pk.q[0];
    reinterpret_cast<int4*>(P + base)[1] = pk.q[1];
  }
}

// ---------------------------------------------------------------------------
// Tiled bf16 WMMA GEMM: out[M,N] = act(A[M,K]bf16 @ diag(rscale)W[K,N]f32 + bias)
// Workgroup: 64x64 tile, 8 waves (4x2), each wave 16x32.
// W tile staged TRANSPOSED (n-major) so B frags are 2x ds_load_b128.
// ---------------------------------------------------------------------------
__global__ __launch_bounds__(256) void gemm_kernel(
    const bf16_t* __restrict__ A, const float* __restrict__ W,
    const float* __restrict__ bias, const float* __restrict__ rscale,
    void* __restrict__ outp, int M, int N, int K, int act, int out_bf16) {
  __shared__ __align__(16) bf16_t As[64][40];   // m x k
  __shared__ __align__(16) bf16_t Wt[64][40];   // n x k (transposed)
  const int tid = threadIdx.x, lane = tid & 31, wid = tid >> 5;
  const int wm = wid & 3, wn = wid >> 2;
  const int bm = blockIdx.x * 64, bn = blockIdx.y * 64;
  v8f acc0 = {}, acc1 = {};
  const int ar = tid >> 2, akc = (tid & 3) * 8;
  const int wr = tid >> 3, wc = (tid & 7) * 8;
  const int m = lane & 15, kh = lane >> 4;

  for (int k0 = 0; k0 < K; k0 += 32) {
    *reinterpret_cast<int4*>(&As[ar][akc]) =
        *reinterpret_cast<const int4*>(A + (size_t)(bm + ar) * K + k0 + akc);
    const float sc = rscale ? rscale[k0 + wr] : 1.f;
    const float* wp = W + (size_t)(k0 + wr) * N + bn + wc;
    float4 w0 = reinterpret_cast<const float4*>(wp)[0];
    float4 w1 = reinterpret_cast<const float4*>(wp)[1];
    Wt[wc + 0][wr] = f2bf(w0.x * sc); Wt[wc + 1][wr] = f2bf(w0.y * sc);
    Wt[wc + 2][wr] = f2bf(w0.z * sc); Wt[wc + 3][wr] = f2bf(w0.w * sc);
    Wt[wc + 4][wr] = f2bf(w1.x * sc); Wt[wc + 5][wr] = f2bf(w1.y * sc);
    Wt[wc + 6][wr] = f2bf(w1.z * sc); Wt[wc + 7][wr] = f2bf(w1.w * sc);
    __syncthreads();

    v16bf af = ld_frag(&As[wm * 16 + m][kh * 16], &As[wm * 16 + m][kh * 16 + 8]);
    v16bf b0 = ld_frag(&Wt[wn * 32 + m][kh * 16], &Wt[wn * 32 + m][kh * 16 + 8]);
    v16bf b1 = ld_frag(&Wt[wn * 32 + 16 + m][kh * 16], &Wt[wn * 32 + 16 + m][kh * 16 + 8]);
    acc0 = __builtin_amdgcn_wmma_f32_16x16x32_bf16(false, af, false, b0, (short)0, acc0, false, false);
    acc1 = __builtin_amdgcn_wmma_f32_16x16x32_bf16(false, af, false, b1, (short)0, acc1, false, false);
    __syncthreads();
  }
  const int lh = lane >> 4;
  #pragma unroll
  for (int sub = 0; sub < 2; ++sub) {
    v8f acc = sub ? acc1 : acc0;
    int n = bn + wn * 32 + sub * 16 + m;
    float bv = bias ? bias[n] : 0.f;
    #pragma unroll
    for (int r = 0; r < 8; ++r) {
      int row = bm + wm * 16 + r + 8 * lh;
      float x = act_apply(acc[r] + bv, act);
      if (out_bf16) reinterpret_cast<bf16_t*>(outp)[(size_t)row * N + n] = f2bf(x);
      else          reinterpret_cast<float*>(outp)[(size_t)row * N + n]  = x;
    }
  }
}

// ---------------------------------------------------------------------------
// Flash attention per (head, q-tile of 16).  8 waves/wg share K/V LDS tiles.
// d=48 padded to 64 for WMMA.  P planes hold pair bias + beta.
// Ks kept [j][d] (transposed B for QK^T); V staged transposed [d][j].
// ---------------------------------------------------------------------------
__global__ __launch_bounds__(256) void attn_kernel(
    const bf16_t* __restrict__ Qb, const bf16_t* __restrict__ Kb,
    const bf16_t* __restrict__ Vb, const bf16_t* __restrict__ P,
    bf16_t* __restrict__ Ob, int S, int blk) {
  __shared__ __align__(16) bf16_t Qs[8][16][80];
  __shared__ __align__(16) bf16_t Ks[32][80];   // j x d (pad d to 64, zeroed)
  __shared__ __align__(16) bf16_t VsT[48][40];  // d x j (transposed)
  __shared__ __align__(16) bf16_t Ps[8][16][48];

  const int tid = threadIdx.x, lane = tid & 31, wid = tid >> 5;
  const int h = blockIdx.y;
  const int qrow0 = (blockIdx.x * 8 + wid) * 16;
  const size_t plane = ((size_t)(blk * 16 + h)) * S * S;
  const float scale = 0.14433756729740643f;  // 1/sqrt(48)

  { // stage own Q tile, zero-pad cols 48..63
    int r = lane >> 1, seg = lane & 1;
    const int4* src = reinterpret_cast<const int4*>(Qb + (size_t)(qrow0 + r) * 768 + h * 48 + seg * 24);
    int4* dst = reinterpret_cast<int4*>(&Qs[wid][r][seg * 24]);
    dst[0] = src[0]; dst[1] = src[1]; dst[2] = src[2];
    int4 zz = {0, 0, 0, 0};
    *reinterpret_cast<int4*>(&Qs[wid][r][48 + seg * 8]) = zz;
  }
  __syncthreads();

  const int m = lane & 15, kh = lane >> 4;
  v16bf aq0 = ld_frag(&Qs[wid][m][kh * 16],      &Qs[wid][m][kh * 16 + 8]);
  v16bf aq1 = ld_frag(&Qs[wid][m][32 + kh * 16], &Qs[wid][m][32 + kh * 16 + 8]);

  v8f o0 = {}, o1 = {}, o2 = {};
  float mprev[8], lsum[8];
  #pragma unroll
  for (int r = 0; r < 8; ++r) { mprev[r] = -1e30f; lsum[r] = 0.f; }

  const int lh = lane >> 4, jl = lane & 15;

  for (int j0c = 0; j0c < S; j0c += 32) {
    __syncthreads();
    if (tid < 192) {  // stage K (row-major) and V (transposed) cooperatively
      int row = tid / 6, seg = tid % 6;
      *reinterpret_cast<int4*>(&Ks[row][seg * 8]) =
          *reinterpret_cast<const int4*>(Kb + (size_t)(j0c + row) * 768 + h * 48 + seg * 8);
      union { int4 q; bf16_t e[8]; } vv;
      vv.q = *reinterpret_cast<const int4*>(Vb + (size_t)(j0c + row) * 768 + h * 48 + seg * 8);
      #pragma unroll
      for (int i2 = 0; i2 < 8; ++i2) VsT[seg * 8 + i2][row] = vv.e[i2];
    } else {          // zero K pad cols 48..63
      int t = tid - 192; int row = t >> 1, seg = t & 1;
      int4 zz = {0, 0, 0, 0};
      *reinterpret_cast<int4*>(&Ks[row][48 + seg * 8]) = zz;
    }
    __syncthreads();

    // scores: two 16x16 tiles (j local 0..15 and 16..31), K = 64 (padded d)
    v8f sA = {}, sB = {};
    {
      v16bf b0 = ld_frag(&Ks[m][kh * 16],      &Ks[m][kh * 16 + 8]);
      v16bf b1 = ld_frag(&Ks[m][32 + kh * 16], &Ks[m][32 + kh * 16 + 8]);
      sA = __builtin_amdgcn_wmma_f32_16x16x32_bf16(false, aq0, false, b0, (short)0, sA, false, false);
      sA = __builtin_amdgcn_wmma_f32_16x16x32_bf16(false, aq1, false, b1, (short)0, sA, false, false);
    }
    {
      v16bf b0 = ld_frag(&Ks[16 + m][kh * 16],      &Ks[16 + m][kh * 16 + 8]);
      v16bf b1 = ld_frag(&Ks[16 + m][32 + kh * 16], &Ks[16 + m][32 + kh * 16 + 8]);
      sB = __builtin_amdgcn_wmma_f32_16x16x32_bf16(false, aq0, false, b0, (short)0, sB, false, false);
      sB = __builtin_amdgcn_wmma_f32_16x16x32_bf16(false, aq1, false, b1, (short)0, sB, false, false);
    }

    float s0[8], s1[8];
    #pragma unroll
    for (int r = 0; r < 8; ++r) {
      int row = qrow0 + r + 8 * lh;
      s0[r] = sA[r] * scale + bf2f(P[plane + (size_t)row * S + j0c + jl]);
      s1[r] = sB[r] * scale + bf2f(P[plane + (size_t)row * S + j0c + 16 + jl]);
    }
    #pragma unroll
    for (int r = 0; r < 8; ++r) {
      float mr = fmaxf(s0[r], s1[r]);
      mr = fmaxf(mr, __shfl_xor(mr, 1, 16));
      mr = fmaxf(mr, __shfl_xor(mr, 2, 16));
      mr = fmaxf(mr, __shfl_xor(mr, 4, 16));
      mr = fmaxf(mr, __shfl_xor(mr, 8, 16));
      float mn = fmaxf(mprev[r], mr);
      float corr = __expf(mprev[r] - mn);
      float p0 = __expf(s0[r] - mn);
      float p1 = __expf(s1[r] - mn);
      float rsum = p0 + p1;
      rsum += __shfl_xor(rsum, 1, 16);
      rsum += __shfl_xor(rsum, 2, 16);
      rsum += __shfl_xor(rsum, 4, 16);
      rsum += __shfl_xor(rsum, 8, 16);
      lsum[r] = lsum[r] * corr + rsum;
      mprev[r] = mn;
      o0[r] *= corr; o1[r] *= corr; o2[r] *= corr;
      Ps[wid][r + 8 * lh][jl]      = f2bf(p0);
      Ps[wid][r + 8 * lh][16 + jl] = f2bf(p1);
    }
    asm volatile("s_wait_dscnt 0" ::: "memory");

    v16bf ap = ld_frag(&Ps[wid][m][kh * 16], &Ps[wid][m][kh * 16 + 8]);
    {
      v16bf bv = ld_frag(&VsT[m][kh * 16], &VsT[m][kh * 16 + 8]);
      o0 = __builtin_amdgcn_wmma_f32_16x16x32_bf16(false, ap, false, bv, (short)0, o0, false, false);
    }
    {
      v16bf bv = ld_frag(&VsT[16 + m][kh * 16], &VsT[16 + m][kh * 16 + 8]);
      o1 = __builtin_amdgcn_wmma_f32_16x16x32_bf16(false, ap, false, bv, (short)0, o1, false, false);
    }
    {
      v16bf bv = ld_frag(&VsT[32 + m][kh * 16], &VsT[32 + m][kh * 16 + 8]);
      o2 = __builtin_amdgcn_wmma_f32_16x16x32_bf16(false, ap, false, bv, (short)0, o2, false, false);
    }
  }

  #pragma unroll
  for (int r = 0; r < 8; ++r) {
    int row = qrow0 + r + 8 * lh;
    float inv = 1.f / lsum[r];
    Ob[(size_t)row * 768 + h * 48 +  0 + jl] = f2bf(o0[r] * inv);
    Ob[(size_t)row * 768 + h * 48 + 16 + jl] = f2bf(o1[r] * inv);
    Ob[(size_t)row * 768 + h * 48 + 32 + jl] = f2bf(o2[r] * inv);
  }
}

// ---------------------------------------------------------------------------
// Elementwise: out = A*B + (C? C:0), all bf16, 8 elems/thread
// ---------------------------------------------------------------------------
__global__ __launch_bounds__(256) void ew_fma_bf16_kernel(
    const bf16_t* __restrict__ A, const bf16_t* __restrict__ B,
    const bf16_t* __restrict__ C, bf16_t* __restrict__ O, int n8) {
  int i = blockIdx.x * 256 + threadIdx.x;
  if (i >= n8) return;
  union U { int4 q; bf16_t e[8]; };
  U ua, ub, uc, uo;
  ua.q = reinterpret_cast<const int4*>(A)[i];
  ub.q = reinterpret_cast<const int4*>(B)[i];
  if (C) uc.q = reinterpret_cast<const int4*>(C)[i];
  #pragma unroll
  for (int k = 0; k < 8; ++k) {
    float c = C ? bf2f(uc.e[k]) : 0.f;
    uo.e[k] = f2bf(bf2f(ua.e[k]) * bf2f(ub.e[k]) + c);
  }
  reinterpret_cast<int4*>(O)[i] = uo.q;
}

// out = SA*X + ST*Y  (SA/ST bf16 gates, X/Y f32), 4 elems/thread
__global__ __launch_bounds__(256) void ew_final_kernel(
    const bf16_t* __restrict__ SA, const float* __restrict__ X,
    const bf16_t* __restrict__ ST, const float* __restrict__ Y,
    float* __restrict__ O, int n4) {
  int i = blockIdx.x * 256 + threadIdx.x;
  if (i >= n4) return;
  float4 x = reinterpret_cast<const float4*>(X)[i];
  float4 y = reinterpret_cast<const float4*>(Y)[i];
  union V { int2 q; bf16_t e[4]; };
  V sa, st;
  sa.q = reinterpret_cast<const int2*>(SA)[i];
  st.q = reinterpret_cast<const int2*>(ST)[i];
  float4 o;
  o.x = bf2f(sa.e[0]) * x.x + bf2f(st.e[0]) * y.x;
  o.y = bf2f(sa.e[1]) * x.y + bf2f(st.e[1]) * y.y;
  o.z = bf2f(sa.e[2]) * x.z + bf2f(st.e[2]) * y.z;
  o.w = bf2f(sa.e[3]) * x.w + bf2f(st.e[3]) * y.w;
  reinterpret_cast<float4*>(O)[i] = o;
}

// ---------------------------------------------------------------------------
extern "C" void kernel_launch(void* const* d_in, const int* in_sizes, int n_in,
                              void* d_out, int out_size, void* d_ws, size_t ws_size,
                              hipStream_t stream) {
  (void)in_sizes; (void)n_in; (void)out_size; (void)ws_size;
  const int S = 1024, CA = 768, CS = 384, NHID = 1536;

  const float* a_in        = (const float*)d_in[0];
  const float* s_in        = (const float*)d_in[1];
  const float* z_in        = (const float*)d_in[2];
  const float* beta_in     = (const float*)d_in[3];
  const float* ln_s_w_attn = (const float*)d_in[4];
  const float* wg_attn     = (const float*)d_in[5];
  const float* wb_attn     = (const float*)d_in[6];
  const float* wq          = (const float*)d_in[7];
  const float* bq          = (const float*)d_in[8];
  const float* wk          = (const float*)d_in[9];
  const float* wv          = (const float*)d_in[10];
  const float* ln_z_w      = (const float*)d_in[11];
  const float* ln_z_b      = (const float*)d_in[12];
  const float* wpb         = (const float*)d_in[13];
  const float* wgate       = (const float*)d_in[14];
  const float* wo          = (const float*)d_in[15];
  const float* wsg_attn    = (const float*)d_in[16];
  const float* bsg_attn    = (const float*)d_in[17];
  const float* ln_s_w_tr   = (const float*)d_in[18];
  const float* wg_tr       = (const float*)d_in[19];
  const float* wb_tr       = (const float*)d_in[20];
  const float* w_swish     = (const float*)d_in[21];
  const float* w_gate2     = (const float*)d_in[22];
  const float* wsg_tr      = (const float*)d_in[23];
  const float* bsg_tr      = (const float*)d_in[24];
  const float* w_out       = (const float*)d_in[25];

  char* ws = (char*)d_ws;
  size_t off = 0;
  auto take = [&](size_t bytes) -> char* {
    char* p = ws + off;
    off = (off + bytes + 255) & ~(size_t)255;
    return p;
  };
  bf16_t* s_n0   = (bf16_t*)take((size_t)S * CS * 2);
  bf16_t* s_bf   = (bf16_t*)take((size_t)S * CS * 2);
  bf16_t* a_n    = (bf16_t*)take((size_t)S * CA * 2);
  bf16_t* Gbuf   = (bf16_t*)take((size_t)S * CA * 2);
  bf16_t* Bt     = (bf16_t*)take((size_t)S * CA * 2);
  bf16_t* ah     = (bf16_t*)take((size_t)S * CA * 2);
  bf16_t* qb     = (bf16_t*)take((size_t)S * CA * 2);
  bf16_t* kb     = (bf16_t*)take((size_t)S * CA * 2);
  bf16_t* vb     = (bf16_t*)take((size_t)S * CA * 2);
  bf16_t* gate   = (bf16_t*)take((size_t)S * CA * 2);
  bf16_t* obuf   = (bf16_t*)take((size_t)S * CA * 2);
  bf16_t* gobuf  = (bf16_t*)take((size_t)S * CA * 2);
  float*  attout = (float*) take((size_t)S * CA * 4);
  bf16_t* sgat   = (bf16_t*)take((size_t)S * CA * 2);
  bf16_t* th     = (bf16_t*)take((size_t)S * CA * 2);
  bf16_t* h1     = (bf16_t*)take((size_t)S * NHID * 2);
  bf16_t* h2     = (bf16_t*)take((size_t)S * NHID * 2);
  bf16_t* hid    = (bf16_t*)take((size_t)S * NHID * 2);
  float*  trbuf  = (float*) take((size_t)S * CA * 4);
  bf16_t* sgtr   = (bf16_t*)take((size_t)S * CA * 2);
  float*  a_cur  = (float*) take((size_t)S * CA * 4);
  bf16_t* WzT    = (bf16_t*)take((size_t)64 * 128 * 2);
  float*  bz     = (float*) take((size_t)64 * 4);
  bf16_t* P      = (bf16_t*)take((size_t)64 * S * S * 2);

  auto gemm = [&](const bf16_t* A, const float* W, const float* bias, const float* rscale,
                  void* out, int M, int N, int K, int act, int obf) {
    gemm_kernel<<<dim3(M / 64, N / 64), 256, 0, stream>>>(A, W, bias, rscale, out, M, N, K, act, obf);
  };

  hipMemcpyAsync(a_cur, a_in, (size_t)S * CA * 4, hipMemcpyDeviceToDevice, stream);
  row_ln_kernel<<<S, 256, 0, stream>>>(s_in, s_n0, CS, 1);
  row_ln_kernel<<<S, 256, 0, stream>>>(s_in, s_bf, CS, 0);
  fuse_z_kernel<<<1, 64, 0, stream>>>(ln_z_w, ln_z_b, wpb, WzT, bz);
  pair_bias_kernel<<<dim3(S / 64, S), 256, 0, stream>>>(z_in, beta_in, WzT, bz, P, S);

  const int nEW  = S * CA / 8 / 256;   // elementwise blocks (c_a)
  const int nEWH = S * NHID / 8 / 256; // elementwise blocks (hidden)
  const int nEWF = S * CA / 4 / 256;

  for (int blk = 0; blk < 4; ++blk) {
    row_ln_kernel<<<S, 256, 0, stream>>>(a_cur, a_n, CA, 1);
    // --- AttentionPairBias ---
    gemm(s_n0, wg_attn + (size_t)blk * CS * CA, nullptr, ln_s_w_attn + (size_t)blk * CS, Gbuf, S, CA, CS, 1, 1);
    gemm(s_n0, wb_attn + (size_t)blk * CS * CA, nullptr, ln_s_w_attn + (size_t)blk * CS, Bt,   S, CA, CS, 0, 1);
    ew_fma_bf16_kernel<<<nEW, 256, 0, stream>>>(Gbuf, a_n, Bt, ah, S * CA / 8);
    gemm(ah, wq + (size_t)blk * CA * CA, bq + (size_t)blk * CA, nullptr, qb, S, CA, CA, 0, 1);
    gemm(ah, wk + (size_t)blk * CA * CA, nullptr, nullptr, kb, S, CA, CA, 0, 1);
    gemm(ah, wv + (size_t)blk * CA * CA, nullptr, nullptr, vb, S, CA, CA, 0, 1);
    gemm(ah, wgate + (size_t)blk * CA * CA, nullptr, nullptr, gate, S, CA, CA, 1, 1);
    attn_kernel<<<dim3(S / 128, 16), 256, 0, stream>>>(qb, kb, vb, P, obuf, S, blk);
    ew_fma_bf16_kernel<<<nEW, 256, 0, stream>>>(gate, obuf, nullptr, gobuf, S * CA / 8);
    gemm(gobuf, wo + (size_t)blk * CA * CA, nullptr, nullptr, attout, S, CA, CA, 0, 0);
    gemm(s_bf, wsg_attn + (size_t)blk * CS * CA, bsg_attn + (size_t)blk * CA, nullptr, sgat, S, CA, CS, 1, 1);
    // --- ConditionedTransitionBlock ---
    gemm(s_n0, wg_tr + (size_t)blk * CS * CA, nullptr, ln_s_w_tr + (size_t)blk * CS, Gbuf, S, CA, CS, 1, 1);
    gemm(s_n0, wb_tr + (size_t)blk * CS * CA, nullptr, ln_s_w_tr + (size_t)blk * CS, Bt,   S, CA, CS, 0, 1);
    ew_fma_bf16_kernel<<<nEW, 256, 0, stream>>>(Gbuf, a_n, Bt, th, S * CA / 8);
    gemm(th, w_swish + (size_t)blk * CA * NHID, nullptr, nullptr, h1, S, NHID, CA, 2, 1);
    gemm(th, w_gate2 + (size_t)blk * CA * NHID, nullptr, nullptr, h2, S, NHID, CA, 0, 1);
    ew_fma_bf16_kernel<<<nEWH, 256, 0, stream>>>(h1, h2, nullptr, hid, S * NHID / 8);
    gemm(hid, w_out + (size_t)blk * NHID * CA, nullptr, nullptr, trbuf, S, CA, NHID, 0, 0);
    gemm(s_bf, wsg_tr + (size_t)blk * CS * CA, bsg_tr + (size_t)blk * CA, nullptr, sgtr, S, CA, CS, 1, 1);
    float* dst = (blk == 3) ? (float*)d_out : a_cur;
    ew_final_kernel<<<nEWF, 256, 0, stream>>>(sgat, attout, sgtr, trbuf, dst, S * CA / 4);
  }
}
